// MultiHeadAttention_18726057410812
// MI455X (gfx1250) — matrix operand order repaired and checked by reading, not compiled, not run on hardware
//
#include <hip/hip_runtime.h>
#include <hip/hip_bf16.h>

// MHA: B=2, S=2048, H=2048, 16 heads x 128. All matmuls via v_wmma_f32_16x16x32_f16.
// Workspace layout (f16): Q[2][16][2048][128] | K same | V^T[2][16][128][2048] | CTX[2][2048][2048]
// Requires ws_size >= 64 MB.

#define HID   2048
#define SEQ   2048
#define NB    2
#define NH    16
#define HD    128
#define MTOT  (NB * SEQ)          // 4096 rows

typedef __attribute__((ext_vector_type(16))) _Float16 v16h;
typedef __attribute__((ext_vector_type(8)))  _Float16 v8h;
typedef __attribute__((ext_vector_type(8)))  float    v8f;
typedef __attribute__((ext_vector_type(4)))  float    v4f;
typedef __attribute__((ext_vector_type(4)))  int      v4i;

union F16Frag { v16h v; v8h h[2]; };

// ---- CDNA5 async global->LDS copy (ASYNCcnt path), guarded for portability ----
#if defined(__AMDGCN__) && __has_builtin(__builtin_amdgcn_global_load_async_to_lds_b128)
#define USE_ASYNC_LDS 1
#else
#define USE_ASYNC_LDS 0
#endif

#if USE_ASYNC_LDS
static __device__ __forceinline__ void async_cp16B(_Float16* lds, const _Float16* g) {
  // 16 bytes per lane, global -> LDS, no VGPR data round-trip
  __builtin_amdgcn_global_load_async_to_lds_b128(
      (__attribute__((address_space(1))) v4i*)(_Float16*)g,
      (__attribute__((address_space(3))) v4i*)lds, 0, 0);
}
#endif

static __device__ __forceinline__ void wait_async0() {
#if __has_builtin(__builtin_amdgcn_s_wait_asynccnt)
  __builtin_amdgcn_s_wait_asynccnt(0);
#else
  asm volatile("s_wait_asynccnt 0" ::: "memory");
#endif
}

static __device__ __forceinline__ v8f wmma_f16(v16h a, v16h b, v8f c) {
  // D = A(16x32 f16) * B(32x16 f16) + C(16x16 f32)
  return __builtin_amdgcn_wmma_f32_16x16x32_f16(false, a, false, b, (short)0, c, false, false);
}

static __device__ __forceinline__ v8f vzero8() {
  v8f z;
#pragma unroll
  for (int i = 0; i < 8; ++i) z[i] = 0.0f;
  return z;
}

static __device__ __forceinline__ v8h cvt8(v4f a, v4f b) {
  v8h r;
#pragma unroll
  for (int i = 0; i < 4; ++i) { r[i] = (_Float16)a[i]; r[i + 4] = (_Float16)b[i]; }
  return r;
}

static __device__ __forceinline__ float redmax16(float x) {
  x = fmaxf(x, __shfl_xor(x, 1, 32));
  x = fmaxf(x, __shfl_xor(x, 2, 32));
  x = fmaxf(x, __shfl_xor(x, 4, 32));
  x = fmaxf(x, __shfl_xor(x, 8, 32));
  return x;
}
static __device__ __forceinline__ float redsum16(float x) {
  x += __shfl_xor(x, 1, 32);
  x += __shfl_xor(x, 2, 32);
  x += __shfl_xor(x, 4, 32);
  x += __shfl_xor(x, 8, 32);
  return x;
}

// ---------------------------------------------------------------------------
// Projection GEMM: Y = A @ W^T + bias.  A:[4096,2048], W:[2048(out),2048(in)].
// MODE 0: A=f32 x, out = f16 Q layout [b][h][s][d], scaled by 1/sqrt(128)
// MODE 1: A=f32 x, out = f16 K layout [b][h][s][d]
// MODE 2: A=f32 x, out = f16 V^T layout [b][h][d][s]
// MODE 3: A=f16 ctx, out = f32 [4096][2048] (d_out)
// Block 256 threads = 8 waves; WG tile 128(M) x 128(N); wave tile 32x64; K-step 32.
// ---------------------------------------------------------------------------
template <int MODE>
__global__ __launch_bounds__(256) void proj_kernel(const void* __restrict__ Ain,
                                                   const float* __restrict__ W,
                                                   const float* __restrict__ bias,
                                                   void* __restrict__ out) {
  __shared__ alignas(16) _Float16 sA[128][40];  // +8 pad: 80B row stride, 16B aligned
  __shared__ alignas(16) _Float16 sB[128][40];

  const int tid  = threadIdx.x;
  const int lane = tid & 31;
  const int w    = tid >> 5;
  const int wm   = w & 3;     // 4 row-groups of 32
  const int wn   = w >> 2;    // 2 col-groups of 64
  const int hi   = lane >> 4; // which 16-lane half
  const int ln   = lane & 15;

  const int Mbase = blockIdx.x * 128;
  const int Nbase = blockIdx.y * 128;

  v8f acc[2][4];
#pragma unroll
  for (int i = 0; i < 2; ++i)
#pragma unroll
    for (int j = 0; j < 4; ++j) acc[i][j] = vzero8();

  // staging addresses (constant across K loop)
  const int rowA = tid >> 1;
  const int cA   = (tid & 1) * 16;
  const int rowB = tid >> 1;
  const int cB   = (tid & 1) * 16;

  for (int kt = 0; kt < HID / 32; ++kt) {
    const int kbase = kt * 32;
    // ---- stage A tile 128x32 (f16 in LDS) ----
    if (MODE == 3) {
      const _Float16* A = (const _Float16*)Ain;
      const _Float16* src = A + (size_t)(Mbase + rowA) * HID + kbase + cA;
#if USE_ASYNC_LDS
      async_cp16B(&sA[rowA][cA], src);
      async_cp16B(&sA[rowA][cA + 8], src + 8);
#else
      const v8h* v = (const v8h*)src;
      *(v8h*)&sA[rowA][cA]     = v[0];
      *(v8h*)&sA[rowA][cA + 8] = v[1];
#endif
      if (kt + 1 < HID / 32) __builtin_prefetch(src + 32, 0, 0);
    } else {
      const float* A = (const float*)Ain;
      const v4f* src = (const v4f*)(A + (size_t)(Mbase + rowA) * HID + kbase + cA);
      *(v8h*)&sA[rowA][cA]     = cvt8(src[0], src[1]);
      *(v8h*)&sA[rowA][cA + 8] = cvt8(src[2], src[3]);
      if (kt + 1 < HID / 32)
        __builtin_prefetch(A + (size_t)(Mbase + rowA) * HID + kbase + 32 + cA, 0, 0);
    }
    // ---- stage B tile 128x32 from W (f16 in LDS, [n][k]) ----
    {
      const v4f* src = (const v4f*)(W + (size_t)(Nbase + rowB) * HID + kbase + cB);
      *(v8h*)&sB[rowB][cB]     = cvt8(src[0], src[1]);
      *(v8h*)&sB[rowB][cB + 8] = cvt8(src[2], src[3]);
      if (kt + 1 < HID / 32)
        __builtin_prefetch(W + (size_t)(Nbase + rowB) * HID + kbase + 32 + cB, 0, 0);
    }
#if USE_ASYNC_LDS
    if (MODE == 3) wait_async0();
#endif
    __syncthreads();

    // ---- fragments + 8 WMMAs ----
    F16Frag af[2], bf[4];
    const int koffA = hi * 8;
#pragma unroll
    for (int sm = 0; sm < 2; ++sm) {
      const int m = wm * 32 + sm * 16 + ln;
      af[sm].h[0] = *(const v8h*)&sA[m][koffA];
      af[sm].h[1] = *(const v8h*)&sA[m][koffA + 16];
    }
    const int koffB = hi * 8;
#pragma unroll
    for (int sn = 0; sn < 4; ++sn) {
      const int n = wn * 64 + sn * 16 + ln;
      bf[sn].h[0] = *(const v8h*)&sB[n][koffB];
      bf[sn].h[1] = *(const v8h*)&sB[n][koffB + 16];
    }
#pragma unroll
    for (int sm = 0; sm < 2; ++sm)
#pragma unroll
      for (int sn = 0; sn < 4; ++sn)
        acc[sm][sn] = wmma_f16(af[sm].v, bf[sn].v, acc[sm][sn]);
    __syncthreads();
  }

  // ---- epilogue ----
  const float scale = 0.08838834764831845f;  // 1/sqrt(128), folded into Q
  float bv[4];
#pragma unroll
  for (int sn = 0; sn < 4; ++sn) bv[sn] = bias[Nbase + wn * 64 + sn * 16 + ln];

#pragma unroll
  for (int sm = 0; sm < 2; ++sm)
#pragma unroll
    for (int sn = 0; sn < 4; ++sn)
#pragma unroll
      for (int r = 0; r < 8; ++r) {
        const int gm = Mbase + wm * 32 + sm * 16 + r + hi * 8;
        const int gn = Nbase + wn * 64 + sn * 16 + ln;
        float val = acc[sm][sn][r] + bv[sn];
        if (MODE == 3) {
          ((float*)out)[(size_t)gm * HID + gn] = val;
        } else {
          if (MODE == 0) val *= scale;
          const int b_ = gm >> 11, s_ = gm & (SEQ - 1);
          const int hh = gn >> 7, dh = gn & (HD - 1);
          _Float16 hv = (_Float16)val;
          if (MODE == 2)  // V^T: [b][h][d][s]
            ((_Float16*)out)[(((size_t)b_ * NH + hh) * HD + dh) * SEQ + s_] = hv;
          else            // Q/K: [b][h][s][d]
            ((_Float16*)out)[(((size_t)b_ * NH + hh) * SEQ + s_) * HD + dh] = hv;
        }
      }
}

// ---------------------------------------------------------------------------
// Flash attention: grid (S/128, B*NH), 256 threads = 8 waves.
// Wave owns 16 query rows; 64-key tiles staged in LDS (async when available);
// online softmax. Scale already folded into Q. 32 WMMAs per softmax pass.
// ---------------------------------------------------------------------------
__global__ __launch_bounds__(256) void attn_kernel(const _Float16* __restrict__ Q,
                                                   const _Float16* __restrict__ K,
                                                   const _Float16* __restrict__ VT,
                                                   _Float16* __restrict__ CTX) {
  __shared__ alignas(16) _Float16 sK[64][136];    // [key][d], 272B rows
  __shared__ alignas(16) _Float16 sV[128][72];    // [d][key], 144B rows
  __shared__ alignas(16) _Float16 sP[8][16][72];  // per-wave P scratch, 16x64

  const int tid  = threadIdx.x;
  const int lane = tid & 31;
  const int w    = tid >> 5;
  const int hi   = lane >> 4;
  const int ln   = lane & 15;

  const int qb = blockIdx.x;
  const int bh = blockIdx.y;
  const int b_ = bh >> 4, h = bh & 15;
  const int qbase = qb * 128 + w * 16;

  const _Float16* Qb  = Q  + (size_t)bh * SEQ * HD;
  const _Float16* Kb  = K  + (size_t)bh * SEQ * HD;
  const _Float16* VTb = VT + (size_t)bh * HD * SEQ;

  // Q fragments (persist in registers): 4 chunks of 16x32 over d=128
  F16Frag qf[4];
  {
    const int m = qbase + ln;
    const int koff = hi * 8;
#pragma unroll
    for (int dc = 0; dc < 4; ++dc) {
      qf[dc].h[0] = *(const v8h*)&Qb[(size_t)m * HD + dc * 32 + koff];
      qf[dc].h[1] = *(const v8h*)&Qb[(size_t)m * HD + dc * 32 + koff + 16];
    }
  }

  v8f o[8];
  float rmax[8], rsum[8];
#pragma unroll
  for (int dc = 0; dc < 8; ++dc) o[dc] = vzero8();
#pragma unroll
  for (int r = 0; r < 8; ++r) { rmax[r] = -3.0e38f; rsum[r] = 0.0f; }

  // staging addresses (constant across tile loop)
  const int rowK = tid >> 2;          // 64 rows, 4 threads/row
  const int cK   = (tid & 3) * 32;    // 32 halfs each
  const int rowV = tid >> 1;          // 128 rows, 2 threads/row
  const int cV   = (tid & 1) * 32;    // 32 halfs each

  const int ktmax = (qb * 128 + 127) >> 6;  // inclusive, 64-key tiles
  for (int kt = 0; kt <= ktmax; ++kt) {
    // ---- stage K tile 64x128 and V^T tile 128x64 ----
    {
      const _Float16* srcK = &Kb[(size_t)(kt * 64 + rowK) * HD + cK];
      const _Float16* srcV = &VTb[(size_t)rowV * SEQ + kt * 64 + cV];
#if USE_ASYNC_LDS
#pragma unroll
      for (int i = 0; i < 4; ++i) async_cp16B(&sK[rowK][cK + i * 8], srcK + i * 8);
#pragma unroll
      for (int i = 0; i < 4; ++i) async_cp16B(&sV[rowV][cV + i * 8], srcV + i * 8);
#else
      const v8h* vK = (const v8h*)srcK;
      const v8h* vV = (const v8h*)srcV;
#pragma unroll
      for (int i = 0; i < 4; ++i) *(v8h*)&sK[rowK][cK + i * 8] = vK[i];
#pragma unroll
      for (int i = 0; i < 4; ++i) *(v8h*)&sV[rowV][cV + i * 8] = vV[i];
#endif
      if (kt < ktmax) {
        __builtin_prefetch(srcK + (size_t)64 * HD, 0, 0);
        __builtin_prefetch(srcV + 64, 0, 0);
      }
    }
#if USE_ASYNC_LDS
    wait_async0();
#endif
    __syncthreads();

    if (kt * 64 <= qbase + 15) {  // wave-uniform: some key in tile is visible
      // ---- S = Q K^T over 64 keys (16 WMMAs) ----
      v8f s[4];
#pragma unroll
      for (int j = 0; j < 4; ++j) s[j] = vzero8();
      F16Frag bk;
#pragma unroll
      for (int j = 0; j < 4; ++j) {
        const int krow = j * 16 + ln;
#pragma unroll
        for (int dc = 0; dc < 4; ++dc) {
          bk.h[0] = *(const v8h*)&sK[krow][dc * 32 + hi * 8];
          bk.h[1] = *(const v8h*)&sK[krow][dc * 32 + hi * 8 + 16];
          s[j] = wmma_f16(qf[dc].v, bk.v, s[j]);
        }
      }
      // ---- causal mask (only needed on/near the diagonal) ----
      if (kt * 64 + 63 > qbase) {
#pragma unroll
        for (int r = 0; r < 8; ++r) {
          const int qg = qbase + r + hi * 8;
#pragma unroll
          for (int j = 0; j < 4; ++j)
            if (kt * 64 + j * 16 + ln > qg) s[j][r] = -3.0e38f;
        }
      }
      // ---- online softmax (row-at-a-time to limit liveness) ----
      float corr[8];
#pragma unroll
      for (int r = 0; r < 8; ++r) {
        float mt = fmaxf(fmaxf(s[0][r], s[1][r]), fmaxf(s[2][r], s[3][r]));
        mt = redmax16(mt);
        const float M = fmaxf(rmax[r], mt);
        corr[r] = __expf(rmax[r] - M);
        rmax[r] = M;
        const int m = r + hi * 8;
        float psum = 0.0f;
#pragma unroll
        for (int j = 0; j < 4; ++j) {
          const float p = __expf(s[j][r] - M);
          psum += p;
          sP[w][m][j * 16 + ln] = (_Float16)p;
        }
        rsum[r] = rsum[r] * corr[r] + redsum16(psum);
      }
#pragma unroll
      for (int dc = 0; dc < 8; ++dc)
#pragma unroll
        for (int r = 0; r < 8; ++r) o[dc][r] *= corr[r];

      asm volatile("s_wait_dscnt 0" ::: "memory");  // wave-local LDS RAW fence

      // ---- P A-fragments (16x64 -> two 16x32 frags) ----
      F16Frag pf[2];
#pragma unroll
      for (int t = 0; t < 2; ++t) {
        pf[t].h[0] = *(const v8h*)&sP[w][ln][t * 32 + hi * 8];
        pf[t].h[1] = *(const v8h*)&sP[w][ln][t * 32 + hi * 8 + 16];
      }
      // ---- O += P V (16 WMMAs across d=128) ----
      F16Frag bv;
#pragma unroll
      for (int dc = 0; dc < 8; ++dc) {
        const int drow = dc * 16 + ln;
#pragma unroll
        for (int t = 0; t < 2; ++t) {
          bv.h[0] = *(const v8h*)&sV[drow][t * 32 + hi * 8];
          bv.h[1] = *(const v8h*)&sV[drow][t * 32 + hi * 8 + 16];
          o[dc] = wmma_f16(pf[t].v, bv.v, o[dc]);
        }
      }
    }
    __syncthreads();
  }

  // ---- normalize + write ctx [b][s][h*128+d] as f16 ----
  float inv[8];
#pragma unroll
  for (int r = 0; r < 8; ++r) inv[r] = 1.0f / rsum[r];
#pragma unroll
  for (int dc = 0; dc < 8; ++dc)
#pragma unroll
    for (int r = 0; r < 8; ++r) {
      const int s_  = qbase + r + hi * 8;
      const int col = h * HD + dc * 16 + ln;
      CTX[((size_t)b_ * SEQ + s_) * HID + col] = (_Float16)(o[dc][r] * inv[r]);
    }
}

extern "C" void kernel_launch(void* const* d_in, const int* in_sizes, int n_in,
                              void* d_out, int out_size, void* d_ws, size_t ws_size,
                              hipStream_t stream) {
  const float* x  = (const float*)d_in[0];
  const float* wq = (const float*)d_in[1];
  const float* bq = (const float*)d_in[2];
  const float* wk = (const float*)d_in[3];
  const float* bk = (const float*)d_in[4];
  const float* wv = (const float*)d_in[5];
  const float* bv = (const float*)d_in[6];
  const float* wo = (const float*)d_in[7];
  const float* bo = (const float*)d_in[8];

  const size_t QKV = (size_t)NB * NH * SEQ * HD;  // 8,388,608 f16 elems (16 MB)
  _Float16* Qw = (_Float16*)d_ws;
  _Float16* Kw = Qw + QKV;
  _Float16* Vw = Kw + QKV;
  _Float16* Cw = Vw + QKV;

  dim3 gproj(MTOT / 128, HID / 128);  // 32 x 16
  dim3 gattn(SEQ / 128, NB * NH);     // 16 x 32

  proj_kernel<0><<<gproj, 256, 0, stream>>>(x,  wq, bq, Qw);
  proj_kernel<1><<<gproj, 256, 0, stream>>>(x,  wk, bk, Kw);
  proj_kernel<2><<<gproj, 256, 0, stream>>>(x,  wv, bv, Vw);
  attn_kernel<<<gattn, 256, 0, stream>>>(Qw, Kw, Vw, Cw);
  proj_kernel<3><<<gproj, 256, 0, stream>>>(Cw, wo, bo, d_out);
}
